// Translator_53034256171585
// MI455X (gfx1250) — matrix operand order repaired and checked by reading, not compiled, not run on hardware
//
#include <hip/hip_runtime.h>
#include <math.h>

// Problem constants (match reference)
#define Bb 2
#define Nn 1024
#define Dd 1024
#define Hh 16
#define Ll 8
#define Mm 4096
#define DHd 64

typedef __attribute__((ext_vector_type(16))) __bf16 v16bf;
typedef __attribute__((ext_vector_type(8)))  __bf16 v8bf;
typedef __attribute__((ext_vector_type(8)))  float  v8f;

__device__ __forceinline__ unsigned short f2bf(float f) {
  union { float f; unsigned u; } x; x.f = f;
  unsigned r = x.u + 0x7FFFu + ((x.u >> 16) & 1u);
  return (unsigned short)(r >> 16);
}

// Build a 16-element bf16 fragment from two 16-byte chunks.
__device__ __forceinline__ v16bf ld_frag(const unsigned short* p0, const unsigned short* p1) {
  union { v16bf v; v8bf h[2]; } u;
  u.h[0] = *reinterpret_cast<const v8bf*>(p0);
  u.h[1] = *reinterpret_cast<const v8bf*>(p1);
  return u.v;
}

// CDNA5 async 16B copy global -> LDS (ASYNCcnt tracked, no VGPR round-trip).
__device__ __forceinline__ void async_cp16(unsigned lds_off, const unsigned short* gptr) {
  asm volatile("global_load_async_to_lds_b128 %0, %1, off"
               :: "v"(lds_off), "v"(gptr)
               : "memory");
}
__device__ __forceinline__ void wait_async0() {
  asm volatile("s_wait_asynccnt 0x0" ::: "memory");
}
__device__ __forceinline__ void wait_async4() {  // allow 4 newest (next stage) in flight
  asm volatile("s_wait_asynccnt 0x4" ::: "memory");
}

// ---------------- elementwise: x + pos_embed -> residual stream (fp32) -----
__global__ void addpos_kernel(const float* __restrict__ x, const float* __restrict__ pos,
                              float* __restrict__ out, int total, int nd) {
  int i = blockIdx.x * blockDim.x + threadIdx.x;
  if (i < total) out[i] = x[i] + pos[i % nd];
}

// ---------------- LayerNorm (fp32 in -> bf16 out) --------------------------
__global__ void layernorm_kernel(const float* __restrict__ x, const float* __restrict__ g,
                                 const float* __restrict__ bta, unsigned short* __restrict__ y) {
  __shared__ float s1[256], s2[256];
  const int row = blockIdx.x, tid = threadIdx.x;
  const float* xr = x + (size_t)row * Dd;
  float vals[4], a = 0.f, b = 0.f;
#pragma unroll
  for (int i = 0; i < 4; ++i) { float v = xr[tid + i * 256]; vals[i] = v; a += v; b += v * v; }
  s1[tid] = a; s2[tid] = b; __syncthreads();
  for (int st = 128; st > 0; st >>= 1) {
    if (tid < st) { s1[tid] += s1[tid + st]; s2[tid] += s2[tid + st]; }
    __syncthreads();
  }
  const float mu = s1[0] * (1.f / Dd);
  const float var = s2[0] * (1.f / Dd) - mu * mu;
  const float rstd = rsqrtf(var + 1e-5f);
#pragma unroll
  for (int i = 0; i < 4; ++i) {
    int c = tid + i * 256;
    y[(size_t)row * Dd + c] = f2bf((vals[i] - mu) * rstd * g[c] + bta[c]);
  }
}

// ---------------- fp32 [K,Nout] -> bf16 [Nout,K] (weight transpose) --------
__global__ void wtrans_kernel(const float* __restrict__ in, unsigned short* __restrict__ out,
                              int K, int Nout) {
  __shared__ float tile[32][33];
  const int n0 = blockIdx.x * 32, k0 = blockIdx.y * 32;
  const int tx = threadIdx.x, ty = threadIdx.y;  // 32 x 8
#pragma unroll
  for (int i = 0; i < 32; i += 8)
    tile[ty + i][tx] = in[(size_t)(k0 + ty + i) * Nout + n0 + tx];
  __syncthreads();
#pragma unroll
  for (int i = 0; i < 32; i += 8)
    out[(size_t)(n0 + ty + i) * K + k0 + tx] = f2bf(tile[tx][ty + i]);
}

// ---------------- double-buffered LDS-tiled WMMA GEMM ----------------------
// A: bf16 [M,K] row-major.  Bt: bf16 [Ncols,K] (transposed weights).
// Block tile 128x128, BK=32 per stage, ping-pong LDS staged via async DMA.
// 256 threads = 8 waves in a 4(M) x 2(N) grid; each wave owns 32x64.
// MODE 0: out bf16 = acc + bias
// MODE 1: out fp32 += acc + bias (residual accumulate)
// MODE 2: out bf16 = gelu(acc + bias)   (exact erf gelu)
#define LS 40  // halves per LDS row (80B): 16B aligned, conflict-free frag reads
template <int MODE>
__global__ void __launch_bounds__(256)
gemm_wmma_kernel(const unsigned short* __restrict__ A, const unsigned short* __restrict__ Bt,
                 const float* __restrict__ bias, void* __restrict__ outp, int Mr, int Nc, int K) {
  __shared__ unsigned short sA[2][128 * LS];
  __shared__ unsigned short sB[2][128 * LS];

  const int tid  = threadIdx.x;
  const int wave = tid >> 5;
  const int lane = tid & 31;
  const int half = lane >> 4;
  const int l16  = lane & 15;
  const int wm   = wave >> 1;  // 0..3 : 32-row strip
  const int wn   = wave & 1;   // 0..1 : 64-col strip

  v8f acc[2][4];
#pragma unroll
  for (int i = 0; i < 2; ++i)
#pragma unroll
    for (int t = 0; t < 4; ++t)
#pragma unroll
      for (int v = 0; v < 8; ++v) acc[i][t][v] = 0.f;

  // staging: thread t copies 32B of one row of A and one row of B per stage
  const int frow  = tid >> 1;  // 0..127
  const int fpart = tid & 1;   // which 32B half of the 64B k-chunk
  const unsigned short* gA = A  + (size_t)(blockIdx.y * 128 + frow) * K + fpart * 16;
  const unsigned short* gB = Bt + (size_t)(blockIdx.x * 128 + frow) * K + fpart * 16;
  unsigned ldsA[2], ldsB[2];
#pragma unroll
  for (int p = 0; p < 2; ++p) {
    ldsA[p] = (unsigned)(size_t)(void*)&sA[p][frow * LS + fpart * 16];
    ldsB[p] = (unsigned)(size_t)(void*)&sB[p][frow * LS + fpart * 16];
  }

  const int nstages = K >> 5;
  // prologue: stage 0 -> buffer 0
#pragma unroll
  for (int i = 0; i < 2; ++i) {
    async_cp16(ldsA[0] + i * 16, gA + i * 8);
    async_cp16(ldsB[0] + i * 16, gB + i * 8);
  }

  for (int s = 0; s < nstages; ++s) {
    const int p = s & 1;
    if (s + 1 < nstages) {
      const int k1 = (s + 1) << 5;
#pragma unroll
      for (int i = 0; i < 2; ++i) {
        async_cp16(ldsA[p ^ 1] + i * 16, gA + k1 + i * 8);
        async_cp16(ldsB[p ^ 1] + i * 16, gB + k1 + i * 8);
      }
      if (s + 2 < nstages) {  // prefetch stage s+2 into L2
        __builtin_prefetch(gA + ((s + 2) << 5), 0, 1);
        __builtin_prefetch(gB + ((s + 2) << 5), 0, 1);
      }
      wait_async4();  // stage s complete; stage s+1 (4 newest) still in flight
    } else {
      wait_async0();
    }
    __syncthreads();

    v16bf af[2];
#pragma unroll
    for (int i = 0; i < 2; ++i) {
      const unsigned short* ar = &sA[p][(wm * 32 + i * 16 + l16) * LS + (half ? 8 : 0)];
      af[i] = ld_frag(ar, ar + 16);
    }
#pragma unroll
    for (int t = 0; t < 4; ++t) {
      const unsigned short* br = &sB[p][(wn * 64 + t * 16 + l16) * LS + half * 8];
      v16bf bf = ld_frag(br, br + 16);
      acc[0][t] = __builtin_amdgcn_wmma_f32_16x16x32_bf16(false, af[0], false, bf,
                                                          (short)0, acc[0][t], false, false);
      acc[1][t] = __builtin_amdgcn_wmma_f32_16x16x32_bf16(false, af[1], false, bf,
                                                          (short)0, acc[1][t], false, false);
    }
    __syncthreads();
  }

  const int rowb = blockIdx.y * 128 + wm * 32;
  const int colb = blockIdx.x * 128 + wn * 64;
#pragma unroll
  for (int i = 0; i < 2; ++i) {
#pragma unroll
    for (int t = 0; t < 4; ++t) {
      const int col = colb + t * 16 + l16;
      const float bv = bias[col];
#pragma unroll
      for (int v = 0; v < 8; ++v) {
        const int row = rowb + i * 16 + v + half * 8;
        float val = acc[i][t][v] + bv;
        const size_t idx = (size_t)row * Nc + col;
        if (MODE == 0) {
          ((unsigned short*)outp)[idx] = f2bf(val);
        } else if (MODE == 1) {
          float* o = (float*)outp;
          o[idx] = o[idx] + val;
        } else {
          val = 0.5f * val * (1.f + erff(val * 0.70710678118654752f));
          ((unsigned short*)outp)[idx] = f2bf(val);
        }
      }
    }
  }
}

// ---------------- split interleaved qkv ------------------------------------
// qkv bf16 [B,N,3D] with col = h*192 + d*3 + sel  ->
//   q,k: [B*H, N, 64] bf16 ; vt: [B*H, 64, N] bf16
__global__ void split_qkv_kernel(const unsigned short* __restrict__ qkv,
                                 unsigned short* __restrict__ q, unsigned short* __restrict__ k,
                                 unsigned short* __restrict__ vt) {
  int i = blockIdx.x * blockDim.x + threadIdx.x;
  if (i >= Bb * Nn * 3 * Dd) return;
  const int c = i % (3 * Dd);
  const int bn = i / (3 * Dd);
  const int n = bn % Nn, b = bn / Nn;
  const int h = c / (3 * DHd);
  const int rem = c % (3 * DHd);
  const int d = rem / 3, sel = rem % 3;
  const int bh = b * Hh + h;
  const unsigned short val = qkv[i];
  if (sel == 0)      q [((size_t)bh * Nn + n) * DHd + d] = val;
  else if (sel == 1) k [((size_t)bh * Nn + n) * DHd + d] = val;
  else               vt[((size_t)bh * DHd + d) * Nn + n] = val;
}

// ---------------- flash attention (causal), one wave per 16 query rows -----
__global__ void __launch_bounds__(32)
attn_kernel(const unsigned short* __restrict__ qb, const unsigned short* __restrict__ kb,
            const unsigned short* __restrict__ vtb, unsigned short* __restrict__ outb) {
  const int qbase = blockIdx.x * 16;
  const int bh = blockIdx.y;
  const int b = bh >> 4, h = bh & 15;
  const int lane = threadIdx.x & 31;
  const int half = lane >> 4, l16 = lane & 15;

  __shared__ unsigned short pLds[16 * 40];  // 16 rows x 32 keys, stride 40 (80B, 16B aligned)

  // Q fragments for K-steps 0 and 32 of DH=64
  const unsigned short* qrow = qb + ((size_t)bh * Nn + qbase + l16) * DHd + half * 8;
  const v16bf qf0 = ld_frag(qrow, qrow + 16);
  const v16bf qf1 = ld_frag(qrow + 32, qrow + 48);

  float rm[8], rl[8];
  v8f o[4];
#pragma unroll
  for (int v = 0; v < 8; ++v) { rm[v] = -1e30f; rl[v] = 0.f; }
#pragma unroll
  for (int t = 0; t < 4; ++t)
#pragma unroll
    for (int v = 0; v < 8; ++v) o[t][v] = 0.f;

  const float scale = 0.125f;  // 1/sqrt(64)

  for (int jb = 0; jb <= qbase; jb += 32) {
    v8f s0, s1;
#pragma unroll
    for (int v = 0; v < 8; ++v) { s0[v] = 0.f; s1[v] = 0.f; }
    const unsigned short* kr0 = kb + ((size_t)bh * Nn + jb + l16) * DHd + half * 8;
    const unsigned short* kr1 = kr0 + 16 * DHd;
    {
      v16bf bk = ld_frag(kr0, kr0 + 16);
      s0 = __builtin_amdgcn_wmma_f32_16x16x32_bf16(false, qf0, false, bk, (short)0, s0, false, false);
      bk = ld_frag(kr0 + 32, kr0 + 48);
      s0 = __builtin_amdgcn_wmma_f32_16x16x32_bf16(false, qf1, false, bk, (short)0, s0, false, false);
      bk = ld_frag(kr1, kr1 + 16);
      s1 = __builtin_amdgcn_wmma_f32_16x16x32_bf16(false, qf0, false, bk, (short)0, s1, false, false);
      bk = ld_frag(kr1 + 32, kr1 + 48);
      s1 = __builtin_amdgcn_wmma_f32_16x16x32_bf16(false, qf1, false, bk, (short)0, s1, false, false);
    }

    // scale + causal mask + online softmax (row = v + half*8 in this C layout)
#pragma unroll
    for (int v = 0; v < 8; ++v) {
      const int rowg = qbase + v + half * 8;
      float a = s0[v] * scale;
      float c = s1[v] * scale;
      if (jb + l16 > rowg) a = -1e30f;
      if (jb + 16 + l16 > rowg) c = -1e30f;
      float t = fmaxf(a, c);
#pragma unroll
      for (int mks = 1; mks < 16; mks <<= 1) t = fmaxf(t, __shfl_xor(t, mks, 32));
      const float nm = fmaxf(rm[v], t);
      const float alpha = __expf(rm[v] - nm);
      a = __expf(a - nm);
      c = __expf(c - nm);
      float rs = a + c;
#pragma unroll
      for (int mks = 1; mks < 16; mks <<= 1) rs += __shfl_xor(rs, mks, 32);
      rl[v] = rl[v] * alpha + rs;
      rm[v] = nm;
#pragma unroll
      for (int t4 = 0; t4 < 4; ++t4) o[t4][v] *= alpha;
      const int prow = v + half * 8;
      pLds[prow * 40 + l16]      = f2bf(a);
      pLds[prow * 40 + 16 + l16] = f2bf(c);
    }
    __syncthreads();

    // reload P as A-fragment (16 rows x 32 keys)
    const unsigned short* pr = pLds + l16 * 40 + (half ? 8 : 0);
    const v16bf pf = ld_frag(pr, pr + 16);

    // O += P @ V   (V pre-transposed: [bh, d, n], contiguous in keys)
#pragma unroll
    for (int t = 0; t < 4; ++t) {
      const unsigned short* vr = vtb + ((size_t)bh * DHd + t * 16 + l16) * Nn + jb + half * 8;
      v16bf bv = ld_frag(vr, vr + 16);
      o[t] = __builtin_amdgcn_wmma_f32_16x16x32_bf16(false, pf, false, bv, (short)0, o[t], false, false);
    }
    __syncthreads();
  }

  // epilogue: normalize, write to [B,N,D] bf16 with D = h*64 + d
#pragma unroll
  for (int t = 0; t < 4; ++t) {
#pragma unroll
    for (int v = 0; v < 8; ++v) {
      const int m = v + half * 8;
      const float val = o[t][v] / rl[v];
      outb[((size_t)(b * Nn + qbase + m)) * Dd + h * DHd + t * 16 + l16] = f2bf(val);
    }
  }
}

// ---------------------------------------------------------------------------
extern "C" void kernel_launch(void* const* d_in, const int* in_sizes, int n_in,
                              void* d_out, int out_size, void* d_ws, size_t ws_size,
                              hipStream_t stream) {
  (void)in_sizes; (void)n_in; (void)out_size; (void)ws_size;

  const float* x        = (const float*)d_in[0];
  const float* pos      = (const float*)d_in[1];
  const float* ln1_g    = (const float*)d_in[2];
  const float* ln1_b    = (const float*)d_in[3];
  const float* qkv_w    = (const float*)d_in[4];
  const float* qkv_b    = (const float*)d_in[5];
  const float* proj_w   = (const float*)d_in[6];
  const float* proj_b   = (const float*)d_in[7];
  const float* ln2_g    = (const float*)d_in[8];
  const float* ln2_b    = (const float*)d_in[9];
  const float* mlp_w1   = (const float*)d_in[10];
  const float* mlp_b1   = (const float*)d_in[11];
  const float* mlp_w2   = (const float*)d_in[12];
  const float* mlp_b2   = (const float*)d_in[13];

  float* hbuf = (float*)d_out;  // residual stream lives in d_out (fp32)

  // ---- workspace carve (all bf16 stored as u16), ~72 MiB
  char* ws = (char*)d_ws;
  size_t off = 0;
  auto carve = [&](size_t bytes) { void* p = ws + off; off += (bytes + 255) & ~(size_t)255; return p; };
  unsigned short* lnbuf  = (unsigned short*)carve((size_t)Bb * Nn * Dd * 2);
  unsigned short* qkvbuf = (unsigned short*)carve((size_t)Bb * Nn * 3 * Dd * 2);
  unsigned short* qbuf   = (unsigned short*)carve((size_t)Bb * Hh * Nn * DHd * 2);
  unsigned short* kbuf   = (unsigned short*)carve((size_t)Bb * Hh * Nn * DHd * 2);
  unsigned short* vtbuf  = (unsigned short*)carve((size_t)Bb * Hh * DHd * Nn * 2);
  unsigned short* attout = (unsigned short*)carve((size_t)Bb * Nn * Dd * 2);
  unsigned short* mlpbuf = (unsigned short*)carve((size_t)Bb * Nn * Mm * 2);
  unsigned short* wq_t   = (unsigned short*)carve((size_t)3 * Dd * Dd * 2);
  unsigned short* wp_t   = (unsigned short*)carve((size_t)Dd * Dd * 2);
  unsigned short* w1_t   = (unsigned short*)carve((size_t)Mm * Dd * 2);
  unsigned short* w2_t   = (unsigned short*)carve((size_t)Dd * Mm * 2);

  const int total = Bb * Nn * Dd;
  addpos_kernel<<<(total + 255) / 256, 256, 0, stream>>>(x, pos, hbuf, total, Nn * Dd);

  const dim3 tb(32, 8);
  for (int l = 0; l < Ll; ++l) {
    wtrans_kernel<<<dim3(3 * Dd / 32, Dd / 32), tb, 0, stream>>>(
        qkv_w + (size_t)l * Dd * 3 * Dd, wq_t, Dd, 3 * Dd);
    wtrans_kernel<<<dim3(Dd / 32, Dd / 32), tb, 0, stream>>>(
        proj_w + (size_t)l * Dd * Dd, wp_t, Dd, Dd);
    wtrans_kernel<<<dim3(Mm / 32, Dd / 32), tb, 0, stream>>>(
        mlp_w1 + (size_t)l * Dd * Mm, w1_t, Dd, Mm);
    wtrans_kernel<<<dim3(Dd / 32, Mm / 32), tb, 0, stream>>>(
        mlp_w2 + (size_t)l * Mm * Dd, w2_t, Mm, Dd);

    // h = LN1(h_in)
    layernorm_kernel<<<Bb * Nn, 256, 0, stream>>>(hbuf, ln1_g + l * Dd, ln1_b + l * Dd, lnbuf);
    // qkv = h @ qkv_w + qkv_b  (bf16 out)
    gemm_wmma_kernel<0><<<dim3(3 * Dd / 128, Bb * Nn / 128), 256, 0, stream>>>(
        lnbuf, wq_t, qkv_b + (size_t)l * 3 * Dd, qkvbuf, Bb * Nn, 3 * Dd, Dd);
    // de-interleave q/k/v (v transposed)
    split_qkv_kernel<<<(Bb * Nn * 3 * Dd + 255) / 256, 256, 0, stream>>>(qkvbuf, qbuf, kbuf, vtbuf);
    // causal flash attention
    attn_kernel<<<dim3(Nn / 16, Bb * Hh), 32, 0, stream>>>(qbuf, kbuf, vtbuf, attout);
    // h_res = h_in + attn @ proj_w + proj_b  (fp32 accumulate into d_out)
    gemm_wmma_kernel<1><<<dim3(Dd / 128, Bb * Nn / 128), 256, 0, stream>>>(
        attout, wp_t, proj_b + (size_t)l * Dd, hbuf, Bb * Nn, Dd, Dd);
    // m = LN2(h_res)
    layernorm_kernel<<<Bb * Nn, 256, 0, stream>>>(hbuf, ln2_g + l * Dd, ln2_b + l * Dd, lnbuf);
    // gelu(m @ w1 + b1)  (bf16 out)
    gemm_wmma_kernel<2><<<dim3(Mm / 128, Bb * Nn / 128), 256, 0, stream>>>(
        lnbuf, w1_t, mlp_b1 + (size_t)l * Mm, mlpbuf, Bb * Nn, Mm, Dd);
    // h = h_res + (.) @ w2 + b2  (fp32 accumulate into d_out)
    gemm_wmma_kernel<1><<<dim3(Dd / 128, Bb * Nn / 128), 256, 0, stream>>>(
        mlpbuf, w2_t, mlp_b2 + (size_t)l * Dd, hbuf, Bb * Nn, Dd, Mm);
  }
}
